// BahdanauAttention_56573309223351
// MI455X (gfx1250) — matrix operand order repaired and checked by reading, not compiled, not run on hardware
//
#include <hip/hip_runtime.h>
#include <hip/hip_bf16.h>
#include <math.h>

// ---------------------------------------------------------------------------
// Bahdanau attention, fused for MI455X (gfx1250).
//   B=32, S=4096, H=K=V=1024
//   out = [context (32*1024) | attention_weights (32*4096)]  (f32)
// ---------------------------------------------------------------------------

typedef __attribute__((ext_vector_type(16))) __bf16 v16bf;
typedef __attribute__((ext_vector_type(8)))  float  v8f;

#define BATCH   32
#define SEQ     4096
#define HDIM    1024
#define M_TILE  64
#define LDS_A_STRIDE 1032   // bf16 elems/row (1024 + 8 pad) -> 2064B, bank-conflict-free

// Single-instruction f32x2 -> packed bf16 (RNE) on CDNA5.
__device__ __forceinline__ unsigned cvt_pk_bf16(float lo, float hi) {
    unsigned r;
    asm("v_cvt_pk_bf16_f32 %0, %1, %2" : "=v"(r) : "v"(lo), "v"(hi));
    return r;
}

// Hardware transcendental tanh (TRANS pipe); v_nop guards the TRANS RAW hazard.
__device__ __forceinline__ float fast_tanh(float x) {
    float r;
    asm volatile("v_tanh_f32 %0, %1\n\tv_nop" : "=v"(r) : "v"(x));
    return r;
}

// XOR-butterfly reduction across the 16 lanes of each half-wave.
// ds_swizzle group-of-32 imm: [4:0]=and_mask(0x1F), [9:5]=or, [14:10]=xor.
__device__ __forceinline__ float xor16_sum(float s) {
#if __has_builtin(__builtin_amdgcn_ds_swizzle)
    s += __int_as_float(__builtin_amdgcn_ds_swizzle(__float_as_int(s), 0x041F)); // xor 1
    s += __int_as_float(__builtin_amdgcn_ds_swizzle(__float_as_int(s), 0x081F)); // xor 2
    s += __int_as_float(__builtin_amdgcn_ds_swizzle(__float_as_int(s), 0x101F)); // xor 4
    s += __int_as_float(__builtin_amdgcn_ds_swizzle(__float_as_int(s), 0x201F)); // xor 8
#else
    s += __shfl_xor(s, 1, 32);
    s += __shfl_xor(s, 2, 32);
    s += __shfl_xor(s, 4, 32);
    s += __shfl_xor(s, 8, 32);
#endif
    return s;
}

// ---------------------------------------------------------------------------
// Kernel 0: Wk_w f32 [1024,1024] -> packed bf16 in workspace (row-major)
// ---------------------------------------------------------------------------
__global__ __launch_bounds__(256) void wk_to_bf16_kernel(
    const float* __restrict__ w, unsigned* __restrict__ out)
{
    int g = blockIdx.x * 256 + threadIdx.x;      // over 262144 float4
    float4 v = ((const float4*)w)[g];
    uint2 p;
    p.x = cvt_pk_bf16(v.x, v.y);
    p.y = cvt_pk_bf16(v.z, v.w);
    ((uint2*)out)[g] = p;
}

// ---------------------------------------------------------------------------
// Kernel 1: q_proj[b][h] = dot(query[b,:], Wq_w[h,:]) + Wq_b[h]   (32x1024)
// ---------------------------------------------------------------------------
__global__ __launch_bounds__(256) void qproj_kernel(
    const float* __restrict__ query, const float* __restrict__ Wq_w,
    const float* __restrict__ Wq_b, float* __restrict__ qproj)
{
    int g = blockIdx.x * 256 + threadIdx.x;       // 32768 outputs
    int b = g >> 10, h = g & 1023;
    const float4* q4 = (const float4*)(query + (long)b * HDIM);
    const float4* w4 = (const float4*)(Wq_w + (long)h * HDIM);
    float acc = 0.0f;
    #pragma unroll 4
    for (int k = 0; k < HDIM / 4; ++k) {
        float4 a = q4[k], w = w4[k];
        acc += a.x * w.x + a.y * w.y + a.z * w.z + a.w * w.w;
    }
    qproj[g] = acc + Wq_b[h];
}

// ---------------------------------------------------------------------------
// Kernel 2: fused  scores[b,s] = sum_n v_w[n] * tanh(qproj[b,n] + Wk_b[n]
//                                  + dot(keys[b,s,:], Wk_w[n,:]))
// bf16 WMMA GEMM, k_proj never materialized. One WG = 64 rows of (b*s).
// ---------------------------------------------------------------------------
__global__ __launch_bounds__(256, 1) void fused_score_kernel(
    const float* __restrict__ keys,            // [B*S, 1024] f32
    const unsigned short* __restrict__ wk16,   // [1024, 1024] bf16 (preconverted)
    const float* __restrict__ Wk_b,            // [1024]
    const float* __restrict__ qproj,           // [32, 1024]
    const float* __restrict__ v_w,             // [1024]
    float* __restrict__ scores)                // [B*S]
{
    extern __shared__ char smem_raw[];
    unsigned short* As = (unsigned short*)smem_raw;                    // bf16 bits
    float* scoreAcc = (float*)(smem_raw + M_TILE * LDS_A_STRIDE * 2);  // [64]

    const int tid  = threadIdx.x;
    const int lane = tid & 31;
    const int hf   = lane >> 4;       // half-wave: 0 -> lanes 0-15, 1 -> 16-31
    const int ln16 = lane & 15;
    const long m0  = (long)blockIdx.x * M_TILE;   // tile never crosses a batch (4096%64==0)
    const int  b   = (int)(m0 >> 12);

    if (tid < M_TILE) scoreAcc[tid] = 0.0f;

    // ---- stage keys tile [64 x 1024] f32 -> bf16 into LDS ----
    const float4* keys4 = (const float4*)(keys + m0 * HDIM);
    #pragma unroll 4
    for (int i = 0; i < 64; ++i) {                  // 16384 float4 / 256 threads
        int idx = tid + i * 256;
        int row = idx >> 8;                         // 256 float4 per row
        int c4  = idx & 255;
        float4 v = keys4[idx];
        uint2 p;
        p.x = cvt_pk_bf16(v.x, v.y);
        p.y = cvt_pk_bf16(v.z, v.w);
        *(uint2*)(As + row * LDS_A_STRIDE + (c4 << 2)) = p;   // 8B aligned
    }
    __syncthreads();

    const int ncol0 = (tid >> 5) * 16;              // this wave's n-tile base

    for (int nc = 0; nc < HDIM; nc += 128) {        // 8 waves cover 128 cols/iter
        const int n = nc + ncol0 + ln16;
        const float qb = qproj[b * HDIM + n] + Wk_b[n];
        const float vw = v_w[n];

        v8f acc[4];
        #pragma unroll
        for (int m = 0; m < 4; ++m) acc[m] = (v8f){0,0,0,0,0,0,0,0};

        // B fragment: 16 consecutive bf16 of Wk row n starting at K = ks + hf*16
        const uint4* Bq = (const uint4*)(wk16 + (long)n * HDIM) + (hf << 1);

        #pragma unroll 2
        for (int ks = 0; ks < HDIM; ks += 32) {
            union { v16bf v; uint4 q[2]; } bfr;
            bfr.q[0] = Bq[0];
            bfr.q[1] = Bq[1];
            Bq += 4;                                // advance 32 K (64B)

            // issue all A-fragment LDS loads before any WMMA
            union { v16bf v; uint4 q[2]; } af[4];
            #pragma unroll
            for (int m = 0; m < 4; ++m) {
                const uint4* arow =
                    (const uint4*)(As + (m * 16 + ln16) * LDS_A_STRIDE);
                af[m].q[0] = arow[(ks >> 3) + 2 * hf];        // K = ks + hf*16 .. +8
                af[m].q[1] = arow[(ks >> 3) + 2 * hf + 1];    // K = ks+hf*16+8 .. +8
            }
            #pragma unroll
            for (int m = 0; m < 4; ++m) {
                acc[m] = __builtin_amdgcn_wmma_f32_16x16x32_bf16(
                    false, af[m].v, false, bfr.v, (short)0, acc[m], false, false);
            }
        }

        // epilogue: tanh + weighted reduce over this 16-col slab
        #pragma unroll
        for (int m = 0; m < 4; ++m) {
            #pragma unroll
            for (int i = 0; i < 8; ++i) {
                float s = xor16_sum(vw * fast_tanh(acc[m][i] + qb));
                if (ln16 == 0) {
                    int row = m * 16 + (hf << 3) + i;   // C layout: VGPR i -> M=i / M=8+i
                    atomicAdd(&scoreAcc[row], s);
                }
            }
        }
    }

    __syncthreads();
    if (tid < M_TILE) scores[m0 + tid] = scoreAcc[tid];
}

// ---------------------------------------------------------------------------
// Kernel 3: masked softmax over S (in place in d_out's attention region)
// ---------------------------------------------------------------------------
__global__ __launch_bounds__(256) void softmax_kernel(
    const int* __restrict__ mask, float* __restrict__ attn)
{
    const int b = blockIdx.x, tid = threadIdx.x;
    __shared__ float red[8];
    float* row = attn + (long)b * SEQ;
    const int* mrow = mask + (long)b * SEQ;

    float vals[16];
    float lmax = -INFINITY;
    #pragma unroll
    for (int i = 0; i < 16; ++i) {
        int s = tid + i * 256;
        float v = (mrow[s] == 0) ? -INFINITY : row[s];
        vals[i] = v;
        lmax = fmaxf(lmax, v);
    }
    for (int off = 16; off > 0; off >>= 1) lmax = fmaxf(lmax, __shfl_xor(lmax, off, 32));
    if ((tid & 31) == 0) red[tid >> 5] = lmax;
    __syncthreads();
    float bmax = red[0];
    #pragma unroll
    for (int i = 1; i < 8; ++i) bmax = fmaxf(bmax, red[i]);
    __syncthreads();

    float lsum = 0.0f;
    #pragma unroll
    for (int i = 0; i < 16; ++i) { float e = __expf(vals[i] - bmax); vals[i] = e; lsum += e; }
    for (int off = 16; off > 0; off >>= 1) lsum += __shfl_xor(lsum, off, 32);
    if ((tid & 31) == 0) red[tid >> 5] = lsum;
    __syncthreads();
    float bsum = 0.0f;
    #pragma unroll
    for (int i = 0; i < 8; ++i) bsum += red[i];
    const float inv = 1.0f / bsum;
    #pragma unroll
    for (int i = 0; i < 16; ++i) row[tid + i * 256] = vals[i] * inv;
}

// ---------------------------------------------------------------------------
// Kernel 4: context[b,d] = sum_s attn[b,s] * values[b,s,d]   (streaming)
// ---------------------------------------------------------------------------
__global__ __launch_bounds__(256) void context_kernel(
    const float* __restrict__ attn, const float* __restrict__ values,
    float* __restrict__ ctx)
{
    const int b = blockIdx.x, dc = blockIdx.y, tid = threadIdx.x;
    const int d = dc * 256 + tid;
    __shared__ float aw[256];
    const float* vb = values + ((long)b * SEQ) * HDIM;
    const float* ab = attn + (long)b * SEQ;
    float acc = 0.0f;
    for (int sc = 0; sc < SEQ; sc += 256) {
        __syncthreads();
        aw[tid] = ab[sc + tid];
        __syncthreads();
        #pragma unroll 4
        for (int s = 0; s < 256; ++s)
            acc += aw[s] * vb[(long)(sc + s) * HDIM + d];
    }
    ctx[b * HDIM + d] = acc;
}

// ---------------------------------------------------------------------------
extern "C" void kernel_launch(void* const* d_in, const int* in_sizes, int n_in,
                              void* d_out, int out_size, void* d_ws, size_t ws_size,
                              hipStream_t stream) {
    const float* query  = (const float*)d_in[0];
    const float* keys   = (const float*)d_in[1];
    const float* values = (const float*)d_in[2];
    const int*   mask   = (const int*)  d_in[3];
    const float* Wq_w   = (const float*)d_in[4];
    const float* Wq_b   = (const float*)d_in[5];
    const float* Wk_w   = (const float*)d_in[6];
    const float* Wk_b   = (const float*)d_in[7];
    const float* v_w    = (const float*)d_in[8];

    float* out   = (float*)d_out;
    float* ctx   = out;                    // [32*1024]
    float* attn  = out + BATCH * HDIM;     // [32*4096] (scores -> softmax in place)

    float*          qproj = (float*)d_ws;                       // 32768 f32 (128KB)
    unsigned short* wk16  = (unsigned short*)((char*)d_ws + BATCH * HDIM * sizeof(float));
                                                                // 1M bf16 (2MB)

    wk_to_bf16_kernel<<<(HDIM * HDIM / 4) / 256, 256, 0, stream>>>(Wk_w, (unsigned*)wk16);
    qproj_kernel<<<(BATCH * HDIM) / 256, 256, 0, stream>>>(query, Wq_w, Wq_b, qproj);

    const size_t lds_bytes = (size_t)M_TILE * LDS_A_STRIDE * 2 + M_TILE * sizeof(float);
    hipFuncSetAttribute((const void*)fused_score_kernel,
                        hipFuncAttributeMaxDynamicSharedMemorySize, (int)lds_bytes);
    fused_score_kernel<<<(BATCH * SEQ) / M_TILE, 256, lds_bytes, stream>>>(
        keys, wk16, Wk_b, qproj, v_w, attn);

    softmax_kernel<<<BATCH, 256, 0, stream>>>(mask, attn);
    context_kernel<<<dim3(BATCH, HDIM / 256), 256, 0, stream>>>(attn, values, ctx);
}
